// UITModel_55611236549324
// MI455X (gfx1250) — matrix operand order repaired and checked by reading, not compiled, not run on hardware
//
#include <hip/hip_runtime.h>

// ---------------------------------------------------------------------------
// UITModel on MI455X (gfx1250): persistent-kernel bf16-WMMA RNN, v2.
//   * ALL GEMM operands bf16 in memory: x pre-converted once; o0/h written as
//     bf16 by the elementwise phase (f32 phase state lives in registers)
//   * per-WG weight slice (224KB) staged into LDS once via TDM tensor_load
//   * hardware-native transcendentals (v_exp/v_sin/v_cos/v_rcp)
//   * 128 timesteps x 2 layers with grid-wide atomic barriers
// ---------------------------------------------------------------------------

typedef __attribute__((ext_vector_type(16))) __bf16 v16bf;
typedef __attribute__((ext_vector_type(8)))  float  v8f;
typedef unsigned int u32;
typedef __attribute__((ext_vector_type(4))) u32 u32x4;
typedef __attribute__((ext_vector_type(8))) int  i32x8;
typedef __attribute__((ext_vector_type(4))) int  i32x4;

#define B_   64
#define T_   128
#define I_   512
#define H_   1024
#define O_   256
#define NWG  64           // one WG per 16 hidden columns
#define PI_F 3.14159265358979323846f

#if defined(__HIP_DEVICE_COMPILE__) && __has_builtin(__builtin_amdgcn_tensor_load_to_lds)
#define HAVE_TDM 1
#endif

// --------------------------- WMMA fragment helpers -------------------------

// A fragment (16x32 bf16, ISA 7.12.2): lane&15 = M row; lanes<16 hold
// K = {k0..k0+7} U {k0+16..k0+23}; lanes>=16 shifted by +8. Pure bf16 source:
// each half-fragment is one 16-byte load.
__device__ __forceinline__ v16bf load_a_frag(const __bf16* p0, int ld0,
                                             const __bf16* p1, int ld1, int split,
                                             int row, int k0, int lane) {
  int koff = (lane >> 4) << 3;
  int ka = k0 + koff;
  int kb = ka + 16;
  const __bf16* pa = (ka < split) ? (p0 + (size_t)row * ld0 + ka)
                                  : (p1 + (size_t)row * ld1 + (ka - split));
  const __bf16* pb = (kb < split) ? (p0 + (size_t)row * ld0 + kb)
                                  : (p1 + (size_t)row * ld1 + (kb - split));
  v16bf a;
  ((uint4*)&a)[0] = *(const uint4*)pa;
  ((uint4*)&a)[1] = *(const uint4*)pb;
  return a;
}

// B fragment (32x16 bf16): lane&15 = N col; K = {k0..k0+7} U {k0+16..k0+23},
// lanes>=16 shifted by +8 (matches A). Weights are K-contiguous so this is 2x ds_load_b128.
__device__ __forceinline__ v16bf load_b_frag(const __bf16* bbase, int bld,
                                             int k0, int lane) {
  const __bf16* bp = bbase + (size_t)(lane & 15) * bld + k0 + ((lane >> 4) << 3);
  v16bf b;
  ((uint4*)&b)[0] = ((const uint4*)bp)[0];
  ((uint4*)&b)[1] = ((const uint4*)(bp + 16))[0];
  return b;
}

// One 16x16 f32 tile, A = [p0 | p1] split at `split`, all bf16.
__device__ __forceinline__ v8f gemm_tile(const __bf16* p0, int ld0,
                                         const __bf16* p1, int ld1, int split,
                                         const __bf16* bbase, int bld,
                                         int m0, int nk, int lane) {
  v8f acc = {0.f, 0.f, 0.f, 0.f, 0.f, 0.f, 0.f, 0.f};
  int row = m0 + (lane & 15);
  for (int kc = 0; kc < nk; ++kc) {
    int k0 = kc << 5;
    v16bf a = load_a_frag(p0, ld0, p1, ld1, split, row, k0, lane);
    v16bf b = load_b_frag(bbase, bld, k0, lane);
    acc = __builtin_amdgcn_wmma_f32_16x16x32_bf16(false, a, false, b,
                                                  (short)0, acc, false, false);
  }
  return acc;
}

// C/D layout: VGPR r -> M = m0 + r + 8*(lane>>4), N = colBase + (lane&15)
__device__ __forceinline__ void store_tile(float* gbuf, int colBase, int m0,
                                           v8f acc, int lane) {
  int col = colBase + (lane & 15);
  int r0  = m0 + ((lane >> 4) << 3);
#pragma unroll
  for (int r = 0; r < 8; ++r) gbuf[(r0 + r) * 32 + col] = acc[r];
}

// --------------------------- TDM weight staging ----------------------------

#ifdef HAVE_TDM
// 2D tile load: `rows` rows of `width` bf16 elements, row stride `stride`
// elements, into LDS offset lds_off. D# per CDNA5 ISA.
__device__ __forceinline__ void tdm_load_2d(u32 lds_off, const __bf16* gptr,
                                            u32 width, u32 rows, u32 stride) {
  unsigned long long ga = (unsigned long long)(const void*)gptr;
  u32x4 g0;
  g0[0] = 1u;                                       // count=1, user mode
  g0[1] = lds_off;                                  // lds_addr (bytes)
  g0[2] = (u32)ga;                                  // global_addr[31:0]
  g0[3] = (u32)((ga >> 32) & 0x01ffffffu) | (2u << 30); // addr[56:32] | type=2
  i32x8 g1;
  g1[0] = (int)(1u << 16);                          // data_size=1 (2 bytes)
  g1[1] = (int)((width & 0xffffu) << 16);           // tensor_dim0[15:0]
  g1[2] = (int)(((width >> 16) & 0xffffu) | ((rows & 0xffffu) << 16)); // dim0 hi | dim1 lo
  g1[3] = (int)(((rows >> 16) & 0xffffu) | ((width & 0xffffu) << 16)); // dim1 hi | tile_dim0
  g1[4] = (int)(rows & 0xffffu);                    // tile_dim1 (tile_dim2=0)
  g1[5] = (int)stride;                              // tensor_dim0_stride[31:0]
  g1[6] = 0;
  g1[7] = 0;
  i32x4 z4 = {0, 0, 0, 0};
#if defined(__clang_major__) && __clang_major__ >= 23
  i32x8 z8 = {0, 0, 0, 0, 0, 0, 0, 0};
  __builtin_amdgcn_tensor_load_to_lds(g0, g1, z4, z4, z8, 0);
#else
  __builtin_amdgcn_tensor_load_to_lds(g0, g1, z4, z4, 0);
#endif
}
#endif

// --------------------------- fast elementwise math -------------------------

__device__ __forceinline__ float fast_tanh(float x) {
  float xc = fminf(fmaxf(x, -15.f), 15.f);
  float e2 = __expf(2.f * xc);
  return (e2 - 1.f) * __builtin_amdgcn_rcpf(e2 + 1.f);
}

// phase cell; hprev/phnew kept in f32 registers by the caller.
__device__ __forceinline__ void phase_cell(float xg, float pg, float hprev,
                                           const float* dg, float eps,
                                           float& out, float& phnew,
                                           float& hc, float& hs) {
  float standard = fast_tanh(xg);
  float sg  = __builtin_amdgcn_rcpf(1.f + __expf(-pg));
  float phi = hprev + sg * PI_F;                 // PERS=1, MULT=1
  const float inv_qg = 4.f / PI_F;               // 1/(pi/4)
  const float qg     = PI_F * 0.25f;
  float q = rintf(phi * inv_qg) * qg;            // round-half-even == jnp.round
  phi = phi + 0.125f * (q - phi);                // QS = 0.125
  const float inv_2pi = 1.f / (2.f * PI_F);
  const float twopi   = 2.f * PI_F;
  phi = phi - floorf(phi * inv_2pi) * twopi;     // jnp.remainder semantics
  float c = 0.f, s = 0.f;
#pragma unroll
  for (int k = 0; k < 4; ++k) {                  // harmonics 1,2,4,8
    float ang = phi * (float)(1 << k);
    c = fmaf(__cosf(ang), dg[k], c);
    s = fmaf(__sinf(ang), dg[k], s);
  }
  out = standard * (1.f + eps * c);
  phnew = phi; hc = c; hs = s;
}

// --------------------------- grid-wide barrier -----------------------------

__device__ __forceinline__ void grid_barrier(unsigned* cnt, unsigned barnum) {
  __threadfence();
  __syncthreads();
  if (threadIdx.x == 0) {
    atomicAdd(cnt, 1u);
    unsigned tgt = barnum * NWG;
    volatile unsigned* vc = cnt;
    while (*vc < tgt) __builtin_amdgcn_s_sleep(8);
  }
  __syncthreads();
  __threadfence();
}

// --------------------------- pack kernels ----------------------------------

// dst[n*K + k] = bf16( k<split ? s0[k][n] : s1[k-split][n] ), src shape [K][N]
__global__ void uit_pack_w(const float* __restrict__ s0, const float* __restrict__ s1,
                           int split, int K, int N, __bf16* __restrict__ dst) {
  int idx = blockIdx.x * blockDim.x + threadIdx.x;
  if (idx >= N * K) return;
  int n = idx / K, k = idx % K;
  float v = (k < split) ? s0[(size_t)k * N + n] : s1[(size_t)(k - split) * N + n];
  dst[idx] = (__bf16)v;
}

// plain fp32 -> bf16 convert, 4 elements/thread
__global__ void uit_pack_x(const float* __restrict__ src, __bf16* __restrict__ dst,
                           int n4) {
  int idx = blockIdx.x * blockDim.x + threadIdx.x;
  if (idx >= n4) return;
  float4 v = ((const float4*)src)[idx];
  __bf16* d = dst + (size_t)idx * 4;
  d[0] = (__bf16)v.x; d[1] = (__bf16)v.y; d[2] = (__bf16)v.z; d[3] = (__bf16)v.w;
}

// --------------------------- persistent RNN kernel -------------------------

__global__ __launch_bounds__(256) void uit_persistent(
    const __bf16* __restrict__ Xbf,  // [B,T,I] bf16
    const float* __restrict__ b0, const float* __restrict__ eps0,
    const float* __restrict__ diag0,
    const float* __restrict__ b1, const float* __restrict__ eps1,
    const float* __restrict__ diag1,
    const __bf16* __restrict__ W0,   // [2H][I+H] bf16, K-contiguous
    const __bf16* __restrict__ W1,   // [2H][2H]  bf16, K-contiguous
    __bf16* __restrict__ h0bf,       // [2][B*H] double-buffered (GEMM operand)
    __bf16* __restrict__ h1bf,       // [2][B*H]
    __bf16* __restrict__ o0bf,       // [B*H]
    __bf16* __restrict__ featsbf,    // [B*3H]
    float* __restrict__ confArr,     // [NWG]
    unsigned* __restrict__ barcnt) {
  extern __shared__ char smem[];
  __bf16* lw0  = (__bf16*)smem;                            // 32 x 1536 bf16
  __bf16* lw1  = (__bf16*)(smem + 32 * 1536 * 2);          // 32 x 2048 bf16
  float*  gbuf = (float*)(smem + 32 * 1536 * 2 + 32 * 2048 * 2); // 64 x 32 f32
  float*  red  = gbuf + 64 * 32;                           // 256 f32

  const int tid     = threadIdx.x;
  const int lane    = tid & 31;
  const int wv      = tid >> 5;          // 8 waves
  const int m0      = (wv & 3) << 4;     // M tile base (4 tiles cover B=64)
  const int colBase = (wv >> 2) << 4;    // 0 = x-gate cols, 16 = phi-gate cols
  const int c0      = blockIdx.x << 4;   // hidden-column base of this WG

  // ---- stage this WG's weight slices into LDS (once, via TDM) ----
#ifdef HAVE_TDM
  if (tid < 32) {
    tdm_load_2d(0u,                        W0 + (size_t)c0 * 1536,        1536u, 16u, 1536u);
    tdm_load_2d(16u * 1536u * 2u,          W0 + (size_t)(H_ + c0) * 1536, 1536u, 16u, 1536u);
    tdm_load_2d(32u * 1536u * 2u,          W1 + (size_t)c0 * 2048,        2048u, 16u, 2048u);
    tdm_load_2d(32u * 1536u * 2u + 16u * 2048u * 2u,
                                           W1 + (size_t)(H_ + c0) * 2048, 2048u, 16u, 2048u);
    __builtin_amdgcn_s_wait_tensorcnt(0);
  }
#else
  for (int i = tid; i < (32 * 1536) / 8; i += 256) {
    int e = i * 8, r = e / 1536, k = e % 1536;
    int src = (r < 16) ? (c0 + r) : (H_ + c0 + (r - 16));
    ((uint4*)lw0)[i] = ((const uint4*)(W0 + (size_t)src * 1536 + k))[0];
  }
  for (int i = tid; i < (32 * 2048) / 8; i += 256) {
    int e = i * 8, r = e / 2048, k = e % 2048;
    int src = (r < 16) ? (c0 + r) : (H_ + c0 + (r - 16));
    ((uint4*)lw1)[i] = ((const uint4*)(W1 + (size_t)src * 2048 + k))[0];
  }
#endif
  __syncthreads();

  // Each thread owns one hidden column (hcol) x 4 batch rows for the whole
  // recurrence: f32 phase state + per-column parameters stay in registers.
  const int ecol = tid & 15;             // local column
  const int hcol = c0 + ecol;            // global hidden index
  const int eb0  = tid >> 4;             // first batch row (rows eb0 + 16j)
  float bx0 = b0[hcol],        bp0 = b0[H_ + hcol];
  float bx1 = b1[hcol],        bp1 = b1[H_ + hcol];
  float ee0 = eps0[hcol],      ee1 = eps1[hcol];
  float dgl0[4], dgl1[4];
#pragma unroll
  for (int k = 0; k < 4; ++k) {
    dgl0[k] = diag0[(size_t)hcol * 4 + k];
    dgl1[k] = diag1[(size_t)hcol * 4 + k];
  }
  float ph0[4] = {0.f, 0.f, 0.f, 0.f};   // layer0 phi state (f32, registers)
  float ph1[4] = {0.f, 0.f, 0.f, 0.f};   // layer1 phi state

  float confLocal = 0.f;
  unsigned bar = 0;
  int par = 0;
  for (int t = 0; t < T_; ++t) {
    // ---------------- layer 0: gates = [x_t | h0] @ W0slice ----------------
    const __bf16* xr  = Xbf + (size_t)t * I_;       // row b at +b*T*I
    const __bf16* h0r = h0bf + (size_t)par * (B_ * H_);
    v8f acc = gemm_tile(xr, T_ * I_, h0r, H_, I_,
                        lw0 + (size_t)colBase * 1536, 1536, m0, (I_ + H_) / 32, lane);
    store_tile(gbuf, colBase, m0, acc, lane);
    __syncthreads();

    __bf16* h0w = h0bf + (size_t)(1 - par) * (B_ * H_);
#pragma unroll
    for (int j = 0; j < 4; ++j) {
      int b = eb0 + (j << 4);
      float xg = gbuf[b * 32 + ecol]      + bx0;
      float pg = gbuf[b * 32 + 16 + ecol] + bp0;
      float out, pn, hc, hs;
      phase_cell(xg, pg, ph0[j], dgl0, ee0, out, pn, hc, hs);
      ph0[j] = pn;
      o0bf[(size_t)b * H_ + hcol] = (__bf16)out;
      h0w[(size_t)b * H_ + hcol]  = (__bf16)pn;
      confLocal += fabsf(hc);
    }
    grid_barrier(barcnt, ++bar);

    // ---------------- layer 1: gates = [o0 | h1] @ W1slice -----------------
    const __bf16* h1r = h1bf + (size_t)par * (B_ * H_);
    acc = gemm_tile(o0bf, H_, h1r, H_, H_,
                    lw1 + (size_t)colBase * 2048, 2048, m0, (2 * H_) / 32, lane);
    store_tile(gbuf, colBase, m0, acc, lane);
    __syncthreads();

    __bf16* h1w = h1bf + (size_t)(1 - par) * (B_ * H_);
#pragma unroll
    for (int j = 0; j < 4; ++j) {
      int b = eb0 + (j << 4);
      float xg = gbuf[b * 32 + ecol]      + bx1;
      float pg = gbuf[b * 32 + 16 + ecol] + bp1;
      float out, pn, hc, hs;
      phase_cell(xg, pg, ph1[j], dgl1, ee1, out, pn, hc, hs);
      ph1[j] = pn;
      h1w[(size_t)b * H_ + hcol] = (__bf16)pn;
      confLocal += fabsf(hc);
      if (t == T_ - 1) {
        featsbf[(size_t)b * 3 * H_ + hcol]           = (__bf16)out;
        featsbf[(size_t)b * 3 * H_ + H_ + hcol]      = (__bf16)hc;
        featsbf[(size_t)b * 3 * H_ + 2 * H_ + hcol]  = (__bf16)hs;
      }
    }
    grid_barrier(barcnt, ++bar);
    par ^= 1;
  }

  // deterministic confidence reduction (one value per WG, no float atomics)
  red[tid] = confLocal;
  __syncthreads();
  for (int s = 128; s > 0; s >>= 1) {
    if (tid < s) red[tid] += red[tid + s];
    __syncthreads();
  }
  if (tid == 0) confArr[blockIdx.x] = red[0];
}

// --------------------------- readout kernels -------------------------------

__global__ __launch_bounds__(256) void uit_readout1(
    const __bf16* __restrict__ featsbf, const __bf16* __restrict__ Wr1,
    const float* __restrict__ b1r, __bf16* __restrict__ hidbf) {
  int tid = threadIdx.x, lane = tid & 31, wv = tid >> 5;
  int m0 = (wv & 3) << 4;
  int n0 = blockIdx.x * 32 + ((wv >> 2) << 4);
  v8f acc = gemm_tile(featsbf, 3 * H_, featsbf, 3 * H_, 3 * H_,
                      Wr1 + (size_t)n0 * (3 * H_), 3 * H_, m0, (3 * H_) / 32, lane);
  int col = n0 + (lane & 15);
  int r0  = m0 + ((lane >> 4) << 3);
  float bb = b1r[col];
#pragma unroll
  for (int r = 0; r < 8; ++r)
    hidbf[(size_t)(r0 + r) * H_ + col] = (__bf16)fast_tanh(acc[r] + bb);
}

__global__ __launch_bounds__(256) void uit_readout2(
    const __bf16* __restrict__ hidbf, const __bf16* __restrict__ Wr2,
    const float* __restrict__ b2r, const float* __restrict__ confArr,
    float* __restrict__ dout) {
  int tid = threadIdx.x, lane = tid & 31, wv = tid >> 5;
  int m0 = (wv & 3) << 4;
  int n0 = blockIdx.x * 32 + ((wv >> 2) << 4);
  v8f acc = gemm_tile(hidbf, H_, hidbf, H_, H_,
                      Wr2 + (size_t)n0 * H_, H_, m0, H_ / 32, lane);
  int col = n0 + (lane & 15);
  int r0  = m0 + ((lane >> 4) << 3);
  float bb = b2r[col];
#pragma unroll
  for (int r = 0; r < 8; ++r)
    dout[(size_t)(r0 + r) * O_ + col] = acc[r] + bb;
  if (blockIdx.x == 0 && tid == 0) {
    float s = 0.f;
    for (int i = 0; i < NWG; ++i) s += confArr[i];
    // avg_conf = 0.5 * sum|h_cos| / (K * B * H * T)
    dout[B_ * O_] = s * 0.5f / (float)(4 * B_ * H_ * T_);
  }
}

// --------------------------- launch ----------------------------------------

extern "C" void kernel_launch(void* const* d_in, const int* in_sizes, int n_in,
                              void* d_out, int out_size, void* d_ws, size_t ws_size,
                              hipStream_t stream) {
  const float* x     = (const float*)d_in[0];
  const float* w_ih0 = (const float*)d_in[1];
  const float* w_hh0 = (const float*)d_in[2];
  const float* b0    = (const float*)d_in[3];
  const float* eps0  = (const float*)d_in[4];
  const float* diag0 = (const float*)d_in[5];
  const float* w_ih1 = (const float*)d_in[6];
  const float* w_hh1 = (const float*)d_in[7];
  const float* b1    = (const float*)d_in[8];
  const float* eps1  = (const float*)d_in[9];
  const float* diag1 = (const float*)d_in[10];
  const float* w1    = (const float*)d_in[11];
  const float* b1r   = (const float*)d_in[12];
  const float* w2    = (const float*)d_in[13];
  const float* b2r   = (const float*)d_in[14];

  char* ws = (char*)d_ws;
  size_t off = 0;
  auto alloc = [&](size_t bytes) {
    char* p = ws + off;
    off = (off + bytes + 255) & ~(size_t)255;
    return p;
  };
  __bf16* W0    = (__bf16*)alloc((size_t)2048 * 1536 * 2);
  __bf16* W1    = (__bf16*)alloc((size_t)2048 * 2048 * 2);
  __bf16* Wr1   = (__bf16*)alloc((size_t)1024 * 3072 * 2);
  __bf16* Wr2   = (__bf16*)alloc((size_t)256 * 1024 * 2);
  __bf16* Xbf   = (__bf16*)alloc((size_t)B_ * T_ * I_ * 2);
  char* zbase = ws + off;
  __bf16* h0bf  = (__bf16*)alloc((size_t)2 * B_ * H_ * 2);
  __bf16* h1bf  = (__bf16*)alloc((size_t)2 * B_ * H_ * 2);
  __bf16* o0bf  = (__bf16*)alloc((size_t)B_ * H_ * 2);
  __bf16* featsbf = (__bf16*)alloc((size_t)B_ * 3 * H_ * 2);
  __bf16* hidbf = (__bf16*)alloc((size_t)B_ * H_ * 2);
  float* confArr = (float*)alloc(NWG * 4);
  unsigned* barcnt = (unsigned*)alloc(256);
  size_t zbytes = (size_t)((ws + off) - zbase);
  hipMemsetAsync(zbase, 0, zbytes, stream);   // zero h-state, conf, barrier

  const int thr = 256;
  uit_pack_w<<<(2048 * 1536 + thr - 1) / thr, thr, 0, stream>>>(w_ih0, w_hh0, 512, 1536, 2048, W0);
  uit_pack_w<<<(2048 * 2048 + thr - 1) / thr, thr, 0, stream>>>(w_ih1, w_hh1, 1024, 2048, 2048, W1);
  uit_pack_w<<<(1024 * 3072 + thr - 1) / thr, thr, 0, stream>>>(w1, w1, 3072, 3072, 1024, Wr1);
  uit_pack_w<<<(256 * 1024 + thr - 1) / thr, thr, 0, stream>>>(w2, w2, 1024, 1024, 256, Wr2);
  int n4 = (B_ * T_ * I_) / 4;
  uit_pack_x<<<(n4 + thr - 1) / thr, thr, 0, stream>>>(x, Xbf, n4);

  // LDS: 96KB (layer0 slice) + 128KB (layer1 slice) + 8KB gates + 1KB reduce
  size_t lds = 32 * 1536 * 2 + 32 * 2048 * 2 + 64 * 32 * 4 + 256 * 4;
  uit_persistent<<<NWG, 256, lds, stream>>>(Xbf, b0, eps0, diag0, b1, eps1, diag1,
                                            W0, W1, h0bf, h1bf, o0bf, featsbf,
                                            confArr, barcnt);
  uit_readout1<<<H_ / 32, 256, 0, stream>>>(featsbf, Wr1, b1r, hidbf);
  uit_readout2<<<O_ / 32, 256, 0, stream>>>(hidbf, Wr2, b2r, confArr, (float*)d_out);
}
